// LSTM_82016695484646
// MI455X (gfx1250) — matrix operand order repaired and checked by reading, not compiled, not run on hardware
//
#include <hip/hip_runtime.h>
#include <hip/hip_bf16.h>

// ---- sizes ----
#define SEQ 8192
#define INF 4096
#define EMB 256
#define HID 512
#define GATES4 2048   // 4*HID
#define NWG 16        // persistent workgroups for recurrent scan

typedef __attribute__((ext_vector_type(16))) __bf16 v16bf;
typedef __attribute__((ext_vector_type(8)))  __bf16 v8bf;
typedef __attribute__((ext_vector_type(8)))  float  v8f;

#define SHUF16(lo, hi) __builtin_shufflevector(lo, hi, 0,1,2,3,4,5,6,7,8,9,10,11,12,13,14,15)

__device__ __forceinline__ float sigm(float x) { return 1.0f / (1.0f + __expf(-x)); }

// ---------------------------------------------------------------------------
// Kernel 0a: stream-convert data (8192x4096 f32) -> bf16.  ~9us at HBM peak.
// ---------------------------------------------------------------------------
__global__ __launch_bounds__(256) void k_cvt_data(const float* __restrict__ src,
                                                  __bf16* __restrict__ dst)
{
  const size_t i = ((size_t)blockIdx.x * 256 + threadIdx.x) * 8;
  float4 f0 = *(const float4*)(src + i);
  float4 f1 = *(const float4*)(src + i + 4);
  v8bf o;
  o[0]=(__bf16)f0.x; o[1]=(__bf16)f0.y; o[2]=(__bf16)f0.z; o[3]=(__bf16)f0.w;
  o[4]=(__bf16)f1.x; o[5]=(__bf16)f1.y; o[6]=(__bf16)f1.z; o[7]=(__bf16)f1.w;
  *(v8bf*)(dst + i) = o;
}

// ---------------------------------------------------------------------------
// Kernel 0b: convert all weights to bf16; zero h; init barrier.
//   idx [0, 1048576)            : W_emb flat copy -> Wembb
//   idx [1048576, 1572864)      : We slices (cols 0..255)  -> Web[4][512][256]
//   idx [1572864, 2621440)      : Wh slices (cols 256..767)-> Whb[4][512][512]
// ---------------------------------------------------------------------------
__global__ __launch_bounds__(256) void k_prep_w(
    const float* __restrict__ Wemb,
    const float* __restrict__ Wf, const float* __restrict__ Wi,
    const float* __restrict__ Ws, const float* __restrict__ Wo,
    __bf16* __restrict__ Wembb, __bf16* __restrict__ Web, __bf16* __restrict__ Whb,
    __bf16* __restrict__ hbufb, unsigned* __restrict__ bar)
{
  const int idx = blockIdx.x * 256 + threadIdx.x;
  if (idx < 1048576) {
    Wembb[idx] = (__bf16)Wemb[idx];
    if (idx < HID) hbufb[idx] = (__bf16)0.0f;
    if (idx < 2)   bar[idx] = 0u;
  } else if (idx < 1572864) {
    const int i2 = idx - 1048576;
    const int g = i2 >> 17;
    const int r = (i2 >> 8) & (HID - 1);
    const int k = i2 & (EMB - 1);
    const float* W = (g == 0) ? Wf : (g == 1) ? Wi : (g == 2) ? Ws : Wo;
    Web[i2] = (__bf16)W[(size_t)r * (EMB + HID) + k];
  } else {
    const int i3 = idx - 1572864;
    const int g = i3 >> 18;
    const int r = (i3 >> 9) & (HID - 1);
    const int k = i3 & (HID - 1);
    const float* W = (g == 0) ? Wf : (g == 1) ? Wi : (g == 2) ? Ws : Wo;
    Whb[i3] = (__bf16)W[(size_t)r * (EMB + HID) + EMB + k];
  }
}

// ---------------------------------------------------------------------------
// Kernel 1: logits = data @ W_emb.T + b_emb  (bf16 WMMA, f32 out)
// Block = 256 threads (8 waves), 16 rows x 256 cols per block.
// A-tile hoisted across the wave's two N-tiles (2 accumulators).
// ---------------------------------------------------------------------------
__global__ __launch_bounds__(256) void k_emb_logits(
    const __bf16* __restrict__ datab, const __bf16* __restrict__ Wembb,
    const float* __restrict__ bemb, float* __restrict__ logits)
{
  const int wave = threadIdx.x >> 5;
  const int lane = threadIdx.x & 31;
  const int mtile = blockIdx.x;            // 0..511
  const int kh = lane >> 4;                // 0/1 : K-half per WMMA layout
  const int row = mtile * 16 + (lane & 15);
  const __bf16* arow = datab + (size_t)row * INF;

  const int col0 = (wave * 2) * 16 + (lane & 15);
  const int col1 = col0 + 16;
  const __bf16* brow0 = Wembb + (size_t)col0 * INF;
  const __bf16* brow1 = Wembb + (size_t)col1 * INF;

  v8f acc0 = {}, acc1 = {};
  for (int k0 = 0; k0 < INF; k0 += 32) {
    __builtin_prefetch(arow + k0 + 256, 0, 1);
    // A 16x32: lane holds K = k0 + kh*16 + j (contiguous 16)
    v8bf alo = *(const v8bf*)(arow + k0 + kh * 16);
    v8bf ahi = *(const v8bf*)(arow + k0 + kh * 16 + 8);
    v16bf A = SHUF16(alo, ahi);
    // B 32x16: lane n=lane&15, K = k0 + kh*16 + j (contiguous 16)
    const __bf16* p0 = brow0 + k0 + kh * 16;
    v16bf B0 = SHUF16(*(const v8bf*)p0, *(const v8bf*)(p0 + 8));
    acc0 = __builtin_amdgcn_wmma_f32_16x16x32_bf16(false, A, false, B0,
                                                   (short)0, acc0, false, false);
    const __bf16* p1 = brow1 + k0 + kh * 16;
    v16bf B1 = SHUF16(*(const v8bf*)p1, *(const v8bf*)(p1 + 8));
    acc1 = __builtin_amdgcn_wmma_f32_16x16x32_bf16(false, A, false, B1,
                                                   (short)0, acc1, false, false);
  }
  const float bias0 = bemb[col0], bias1 = bemb[col1];
  #pragma unroll
  for (int r = 0; r < 8; ++r) {
    const int mrow = mtile * 16 + r + kh * 8;  // lanes 16-31 hold M=8..15
    logits[(size_t)mrow * EMB + col0] = acc0[r] + bias0;
    logits[(size_t)mrow * EMB + col1] = acc1[r] + bias1;
  }
}

// ---------------------------------------------------------------------------
// Kernel 2: row softmax (256 cols) -> bf16 emb
// ---------------------------------------------------------------------------
__global__ __launch_bounds__(256) void k_softmax(const float* __restrict__ logits,
                                                 __bf16* __restrict__ emb)
{
  __shared__ float red[256];
  const int row = blockIdx.x, t = threadIdx.x;
  const float v = logits[(size_t)row * EMB + t];
  red[t] = v; __syncthreads();
  for (int s = 128; s > 0; s >>= 1) { if (t < s) red[t] = fmaxf(red[t], red[t + s]); __syncthreads(); }
  const float mx = red[0]; __syncthreads();
  const float e = __expf(v - mx);
  red[t] = e; __syncthreads();
  for (int s = 128; s > 0; s >>= 1) { if (t < s) red[t] += red[t + s]; __syncthreads(); }
  emb[(size_t)row * EMB + t] = (__bf16)(e / red[0]);
}

// ---------------------------------------------------------------------------
// Kernel 3: X = emb @ [We_f|We_i|We_s|We_o].T + bias -> f32 [SEQ x 2048]
// grid (512, 8), block 256; each wave two adjacent 16x16 tiles, K=256.
// ---------------------------------------------------------------------------
__global__ __launch_bounds__(256) void k_gates(
    const __bf16* __restrict__ emb, const __bf16* __restrict__ Web,
    const float* __restrict__ bf_, const float* __restrict__ bi_,
    const float* __restrict__ bs_, const float* __restrict__ bo_,
    float* __restrict__ X)
{
  const int wave = threadIdx.x >> 5;
  const int lane = threadIdx.x & 31;
  const int mtile = blockIdx.x;
  const int kh = lane >> 4;
  const int row = mtile * 16 + (lane & 15);
  const __bf16* arow = emb + (size_t)row * EMB;

  const int ntile0 = blockIdx.y * 16 + wave * 2;       // even, 0..126
  const int gate = ntile0 >> 5;                        // uniform for both tiles
  const int col0 = ntile0 * 16 + (lane & 15);
  const int col1 = col0 + 16;
  const int j0 = col0 & (HID - 1), j1 = col1 & (HID - 1);
  const __bf16* Wg = Web + ((size_t)gate << 17);       // [512][256] bf16
  const __bf16* brow0 = Wg + (size_t)j0 * EMB;
  const __bf16* brow1 = Wg + (size_t)j1 * EMB;

  v8f acc0 = {}, acc1 = {};
  #pragma unroll
  for (int k0 = 0; k0 < EMB; k0 += 32) {
    v8bf alo = *(const v8bf*)(arow + k0 + kh * 16);
    v8bf ahi = *(const v8bf*)(arow + k0 + kh * 16 + 8);
    v16bf A = SHUF16(alo, ahi);
    const __bf16* p0 = brow0 + k0 + kh * 16;
    v16bf B0 = SHUF16(*(const v8bf*)p0, *(const v8bf*)(p0 + 8));
    acc0 = __builtin_amdgcn_wmma_f32_16x16x32_bf16(false, A, false, B0,
                                                   (short)0, acc0, false, false);
    const __bf16* p1 = brow1 + k0 + kh * 16;
    v16bf B1 = SHUF16(*(const v8bf*)p1, *(const v8bf*)(p1 + 8));
    acc1 = __builtin_amdgcn_wmma_f32_16x16x32_bf16(false, A, false, B1,
                                                   (short)0, acc1, false, false);
  }
  const float* bg = (gate == 0) ? bf_ : (gate == 1) ? bi_ : (gate == 2) ? bs_ : bo_;
  const float bias0 = bg[j0], bias1 = bg[j1];
  #pragma unroll
  for (int r = 0; r < 8; ++r) {
    const int mrow = mtile * 16 + r + kh * 8;
    X[(size_t)mrow * GATES4 + col0] = acc0[r] + bias0;
    X[(size_t)mrow * GATES4 + col1] = acc1[r] + bias1;
  }
}

// ---------------------------------------------------------------------------
// Kernel 4: recurrent scan, WMMA matvec with VGPR-resident weights.
// 16 persistent WGs x 256 threads. WG w owns h-slice [w*32, w*32+32) for all
// 4 gates = 128 weight rows. Each wave owns 16 rows, pre-packed as 16 WMMA
// B-tiles (16 x v16bf = 128 VGPRs). Per step: stage h (bf16, 1KB LDS),
// 16 chained WMMAs (A = h broadcast into all 16 rows), reduce, elementwise,
// global sense-reversing barrier.
// ---------------------------------------------------------------------------
__global__ __launch_bounds__(256) void k_lstm(
    const __bf16* __restrict__ Whb, const float* __restrict__ X,
    __bf16* __restrict__ hbufb, unsigned* __restrict__ bar)
{
  __shared__ __align__(16) __bf16 hl[HID];   // 1 KB
  __shared__ float dots[128];

  const int w = blockIdx.x, t = threadIdx.x;
  const int wave = t >> 5, lane = t & 31;
  const int kh = lane >> 4;

  // Pre-pack this wave's 16 weight rows as WMMA B-tiles (VGPR-resident).
  // rloc = wave*16 + n  (n = lane&15): gate g = rloc>>5, row j = w*32+(rloc&31)
  const int rloc = wave * 16 + (lane & 15);
  const int g = rloc >> 5;
  const int jrow = w * 32 + (rloc & 31);
  const __bf16* wrow = Whb + ((size_t)g << 18) + (size_t)jrow * HID;
  v16bf B[16];
  #pragma unroll
  for (int c = 0; c < 16; ++c) {
    const __bf16* p = wrow + c * 32 + kh * 16;
    B[c] = SHUF16(*(const v8bf*)p, *(const v8bf*)(p + 8));
  }

  float c_state = 0.0f;
  const int jmine = w * 32 + t;    // valid for t < 32
  unsigned sense = 0;
  __syncthreads();

  for (int step = 0; step < SEQ; ++step) {
    if (t < 64) *(uint4*)(&hl[t * 8]) = *(const uint4*)(&hbufb[t * 8]);
    __syncthreads();

    // A-tile: every lane loads the h K-slice for its kh -> all 16 A rows == h,
    // so D row 0 (acc[0], lanes 0..15) = the 16 dot products for this wave.
    v8f acc = {};
    #pragma unroll
    for (int c = 0; c < 16; ++c) {
      const __bf16* p = &hl[c * 32 + kh * 16];
      v16bf A = SHUF16(*(const v8bf*)p, *(const v8bf*)(p + 8));
      acc = __builtin_amdgcn_wmma_f32_16x16x32_bf16(false, A, false, B[c],
                                                    (short)0, acc, false, false);
    }
    if (lane < 16) dots[wave * 16 + lane] = acc[0];
    __syncthreads();

    if (t < 32) {
      const float* xr = X + (size_t)step * GATES4;
      const float f  = sigm(xr[0 * HID + jmine] + dots[0 * 32 + t]);
      const float ig = sigm(xr[1 * HID + jmine] + dots[1 * 32 + t]);
      const float sg = tanhf(xr[2 * HID + jmine] + dots[2 * 32 + t]);
      const float o  = sigm(xr[3 * HID + jmine] + dots[3 * 32 + t]);
      c_state = f * c_state + ig * sg;
      hbufb[jmine] = (__bf16)(tanhf(c_state) * o);
    }
    __threadfence();
    __syncthreads();
    sense ^= 1u;
    if (t == 0) {
      unsigned a = __hip_atomic_fetch_add(&bar[0], 1u, __ATOMIC_ACQ_REL,
                                          __HIP_MEMORY_SCOPE_AGENT);
      if (a == NWG - 1) {
        __hip_atomic_store(&bar[0], 0u, __ATOMIC_RELAXED, __HIP_MEMORY_SCOPE_AGENT);
        __hip_atomic_store(&bar[1], sense, __ATOMIC_RELEASE, __HIP_MEMORY_SCOPE_AGENT);
      } else {
        while (__hip_atomic_load(&bar[1], __ATOMIC_ACQUIRE,
                                 __HIP_MEMORY_SCOPE_AGENT) != sense) {
          __builtin_amdgcn_s_sleep(1);
        }
      }
    }
    __syncthreads();
  }
}

// ---------------------------------------------------------------------------
// Kernel 5: out = sigmoid(h @ W_out.T + b_out)   (2 outputs)
// ---------------------------------------------------------------------------
__global__ __launch_bounds__(512) void k_out(const __bf16* __restrict__ hbufb,
                                             const float* __restrict__ Wout,
                                             const float* __restrict__ bout,
                                             float* __restrict__ out)
{
  __shared__ float r0[512], r1[512];
  const int t = threadIdx.x;
  const float h = (float)hbufb[t];
  r0[t] = h * Wout[t];
  r1[t] = h * Wout[HID + t];
  __syncthreads();
  for (int s = 256; s > 0; s >>= 1) {
    if (t < s) { r0[t] += r0[t + s]; r1[t] += r1[t + s]; }
    __syncthreads();
  }
  if (t == 0) {
    out[0] = sigm(r0[0] + bout[0]);
    out[1] = sigm(r1[0] + bout[1]);
  }
}

// ---------------------------------------------------------------------------
extern "C" void kernel_launch(void* const* d_in, const int* in_sizes, int n_in,
                              void* d_out, int out_size, void* d_ws, size_t ws_size,
                              hipStream_t stream)
{
  const float* data = (const float*)d_in[0];
  const float* Wemb = (const float*)d_in[1];
  const float* bemb = (const float*)d_in[2];
  const float* Wf   = (const float*)d_in[3];
  const float* bf_  = (const float*)d_in[4];
  const float* Wi   = (const float*)d_in[5];
  const float* bi_  = (const float*)d_in[6];
  const float* Ws   = (const float*)d_in[7];
  const float* bs_  = (const float*)d_in[8];
  const float* Wo   = (const float*)d_in[9];
  const float* bo_  = (const float*)d_in[10];
  const float* Wout = (const float*)d_in[11];
  const float* bout = (const float*)d_in[12];
  float* out = (float*)d_out;

  // Workspace layout (bytes). datab (67MB) is dead after k_emb_logits and is
  // aliased by X (64MB), which is written later in the stream order.
  char* ws = (char*)d_ws;
  __bf16*   datab  = (__bf16*)  (ws);                 // [0, 67108864)
  float*    X      = (float*)   (ws);                 // [0, 67108864) (aliases datab)
  float*    logits = (float*)   (ws + 67108864);      // 8 MB
  __bf16*   emb    = (__bf16*)  (ws + 75497472);      // 4 MB
  __bf16*   Whb    = (__bf16*)  (ws + 79691776);      // 2 MB
  __bf16*   Web    = (__bf16*)  (ws + 81788928);      // 1 MB
  __bf16*   Wembb  = (__bf16*)  (ws + 82837504);      // 2 MB
  __bf16*   hbufb  = (__bf16*)  (ws + 84934656);      // 1 KB
  unsigned* bar    = (unsigned*)(ws + 84935680);      // 8 B

  k_cvt_data  <<<16384, 256, 0, stream>>>(data, datab);
  k_prep_w    <<<10240, 256, 0, stream>>>(Wemb, Wf, Wi, Ws, Wo,
                                          Wembb, Web, Whb, hbufb, bar);
  k_emb_logits<<<512, 256, 0, stream>>>(datab, Wembb, bemb, logits);
  k_softmax   <<<8192, 256, 0, stream>>>(logits, emb);
  k_gates     <<<dim3(512, 8), 256, 0, stream>>>(emb, Web, bf_, bi_, bs_, bo_, X);
  k_lstm      <<<NWG, 256, 0, stream>>>(Whb, X, hbufb, bar);
  k_out       <<<1, 512, 0, stream>>>(hbufb, Wout, bout, out);
}
